// TransformerTranslation_51625506898377
// MI455X (gfx1250) — matrix operand order repaired and checked by reading, not compiled, not run on hardware
//
#include <hip/hip_runtime.h>

// ---------------------------------------------------------------------------
// Types
// ---------------------------------------------------------------------------
typedef __bf16 bf16_t;
typedef __attribute__((ext_vector_type(16))) __bf16 v16bf;
typedef __attribute__((ext_vector_type(8)))  __bf16 v8bf;
typedef __attribute__((ext_vector_type(8)))  float  v8f;

#define BB    256
#define TT    10
#define DD    64
#define HH    4
#define HSZ   16
#define FFD   256
#define MROWS (BB * TT)   // 2560
#define VOUTD 32000

// ---------------------------------------------------------------------------
// Weight prep: f32 [K,N] -> bf16 [N,K]
// ---------------------------------------------------------------------------
__global__ void k_conv_transpose(const float* __restrict__ in, bf16_t* __restrict__ out,
                                 int K, int N) {
  int i = blockIdx.x * blockDim.x + threadIdx.x;
  if (i >= K * N) return;
  int k = i / N, n = i % N;
  out[(size_t)n * K + k] = (bf16_t)in[i];
}

// per-head attn weight f32 [H=4][D=64][HS=16] -> packed bf16 [N=64][K=64]
// output column n = h*16 + s ; out[n][d] = in[h][d][s]
__global__ void k_conv_attn_w(const float* __restrict__ in, bf16_t* __restrict__ out) {
  int i = blockIdx.x * blockDim.x + threadIdx.x;   // 0..4095 ; i = n*64 + d
  if (i >= 64 * 64) return;
  int n = i >> 6, d = i & 63;
  int h = n >> 4, s = n & 15;
  out[i] = (bf16_t)in[h * (DD * HSZ) + d * HSZ + s];
}

// ---------------------------------------------------------------------------
// Embeddings
// ---------------------------------------------------------------------------
__global__ void k_embed_enc(const int* __restrict__ idx, const float* __restrict__ tok,
                            const float* __restrict__ pos, float* __restrict__ x) {
  int i = blockIdx.x * blockDim.x + threadIdx.x;   // MROWS*64
  if (i >= MROWS * DD) return;
  int d = i & 63, bt = i >> 6, t = bt % TT;
  x[i] = tok[(size_t)idx[bt] * DD + d] + pos[t * DD + d];
}

__global__ void k_embed_dec(const int* __restrict__ tgt, const float* __restrict__ tok,
                            const float* __restrict__ pos, float* __restrict__ y) {
  int i = blockIdx.x * blockDim.x + threadIdx.x;
  if (i >= MROWS * DD) return;
  int d = i & 63, bt = i >> 6, t = bt % TT, b = bt / TT;
  int token = (t == 0) ? 0 : tgt[b * TT + t - 1];   // SOS = 0
  y[i] = tok[(size_t)token * DD + d] + pos[t * DD + d];
}

// ---------------------------------------------------------------------------
// LayerNorm over D=64, output bf16 (feeds WMMA A)
// ---------------------------------------------------------------------------
__global__ void k_ln_bf16(const float* __restrict__ x, const float* __restrict__ g,
                          const float* __restrict__ b, bf16_t* __restrict__ out) {
  int r = blockIdx.x * blockDim.x + threadIdx.x;
  if (r >= MROWS) return;
  const float* xr = x + (size_t)r * DD;
  float m = 0.f;
  #pragma unroll
  for (int i = 0; i < DD; ++i) m += xr[i];
  m *= (1.0f / DD);
  float var = 0.f;
  #pragma unroll
  for (int i = 0; i < DD; ++i) { float d = xr[i] - m; var += d * d; }
  var *= (1.0f / DD);
  float rs = rsqrtf(var + 1e-5f);
  bf16_t* o = out + (size_t)r * DD;
  #pragma unroll
  for (int i = 0; i < DD; ++i) o[i] = (bf16_t)((xr[i] - m) * rs * g[i] + b[i]);
}

// ---------------------------------------------------------------------------
// Attention core: one thread per (b,h,t). T=10, HS=16. scale = D^-0.5 = 0.125
// q,k,v f32 [M,64] with col = h*16+s ; writes f32 + bf16 outputs.
// ---------------------------------------------------------------------------
__global__ void k_attn(const float* __restrict__ q, const float* __restrict__ k,
                       const float* __restrict__ v, float* __restrict__ outf,
                       bf16_t* __restrict__ outbf, int causal) {
  int tid = blockIdx.x * blockDim.x + threadIdx.x;
  if (tid >= BB * HH * TT) return;
  int t = tid % TT, h = (tid / TT) % HH, b = tid / (TT * HH);
  const float scale = 0.125f;
  const float* qrow = q + ((size_t)(b * TT + t) * DD + h * HSZ);
  float qs[HSZ];
  #pragma unroll
  for (int s = 0; s < HSZ; ++s) qs[s] = qrow[s];

  int lim = causal ? t : (TT - 1);
  float sc[TT];
  float mx = -1e30f;
  for (int u = 0; u <= lim; ++u) {
    const float* krow = k + ((size_t)(b * TT + u) * DD + h * HSZ);
    float d = 0.f;
    #pragma unroll
    for (int s = 0; s < HSZ; ++s) d += qs[s] * krow[s];
    d *= scale;
    sc[u] = d;
    mx = fmaxf(mx, d);
  }
  float sum = 0.f;
  for (int u = 0; u <= lim; ++u) { float e = __expf(sc[u] - mx); sc[u] = e; sum += e; }
  float inv = 1.f / sum;
  float o[HSZ];
  #pragma unroll
  for (int s = 0; s < HSZ; ++s) o[s] = 0.f;
  for (int u = 0; u <= lim; ++u) {
    float p = sc[u] * inv;
    const float* vrow = v + ((size_t)(b * TT + u) * DD + h * HSZ);
    #pragma unroll
    for (int s = 0; s < HSZ; ++s) o[s] += p * vrow[s];
  }
  size_t obase = (size_t)(b * TT + t) * DD + h * HSZ;
  #pragma unroll
  for (int s = 0; s < HSZ; ++s) { outf[obase + s] = o[s]; outbf[obase + s] = (bf16_t)o[s]; }
}

// ---------------------------------------------------------------------------
// WMMA GEMM: C[M,N] = A[M,K](bf16) x Bt[N,K](bf16)^T (+bias)(+res)(relu)
// one wave per 16x64 output tile, K-loop step 32, v_wmma_f32_16x16x32_bf16
// ---------------------------------------------------------------------------
__global__ __launch_bounds__(32) void k_gemm(
    const bf16_t* __restrict__ A, const bf16_t* __restrict__ Bt,
    const float* __restrict__ bias, const float* __restrict__ res,
    float* __restrict__ C, bf16_t* __restrict__ Cbf,
    int M, int N, int K, int relu) {
  const int lane   = threadIdx.x;
  const int lane16 = lane & 15;
  const int half   = lane >> 4;
  const int m0 = blockIdx.x << 4;
  const int n0 = blockIdx.y << 6;

  v8f acc[4] = {};

  // A layout (16-bit 16x32): lane row = lane%16 ; elems e -> K=kt+half*16+e
  // (one contiguous 32B run)
  const bf16_t* Arow = A + (size_t)(m0 + lane16) * K;
  for (int kt = 0; kt < K; kt += 32) {
    v16bf a;
    {
      a = *(const v16bf*)(Arow + kt + half * 16);
    }
    // B layout (16-bit 32x16): lane col = lane%16 ; elems e -> K = kt + half*16 + e
    // with Bt stored [N,K] this is one contiguous 32B run per lane.
    #pragma unroll
    for (int nt = 0; nt < 4; ++nt) {
      const bf16_t* bp = Bt + (size_t)(n0 + nt * 16 + lane16) * K + kt + half * 16;
      v16bf bfrag = *(const v16bf*)bp;
      acc[nt] = __builtin_amdgcn_wmma_f32_16x16x32_bf16(
          false, a, false, bfrag, (short)0, acc[nt], false, false);
    }
  }
  // C/D layout: VGPR j -> row m0 + j + 8*half, col n0 + nt*16 + lane%16
  #pragma unroll
  for (int nt = 0; nt < 4; ++nt) {
    const int c = n0 + nt * 16 + lane16;
    const float bv = bias ? bias[c] : 0.0f;
    #pragma unroll
    for (int j = 0; j < 8; ++j) {
      const int r = m0 + j + half * 8;
      float val = acc[nt][j] + bv;
      if (res)  val += res[(size_t)r * N + c];
      if (relu) val = val > 0.f ? val : 0.f;
      C[(size_t)r * N + c] = val;
      if (Cbf) Cbf[(size_t)r * N + c] = (bf16_t)val;
    }
  }
}

// ---------------------------------------------------------------------------
// Host orchestration
// ---------------------------------------------------------------------------
extern "C" void kernel_launch(void* const* d_in, const int* in_sizes, int n_in,
                              void* d_out, int out_size, void* d_ws, size_t ws_size,
                              hipStream_t stream) {
  (void)in_sizes; (void)n_in; (void)out_size; (void)ws_size;
  const int* idx = (const int*)d_in[0];
  const int* tgt = (const int*)d_in[1];
  auto F = [&](int i) -> const float* { return (const float*)d_in[i]; };

  // ---- workspace carve (256B aligned) ----
  char* base = (char*)d_ws;
  size_t off = 0;
  auto alloc = [&](size_t bytes) -> void* {
    void* p = base + off;
    off += (bytes + 255) & ~(size_t)255;
    return p;
  };
  bf16_t* lmwt = (bf16_t*)alloc((size_t)VOUTD * DD * 2);
  bf16_t *qwt[6], *kwt[6], *vwt[6], *pwt[6];
  for (int i = 0; i < 6; ++i) {
    qwt[i] = (bf16_t*)alloc(DD * DD * 2); kwt[i] = (bf16_t*)alloc(DD * DD * 2);
    vwt[i] = (bf16_t*)alloc(DD * DD * 2); pwt[i] = (bf16_t*)alloc(DD * DD * 2);
  }
  bf16_t *w1t[4], *w2t[4];
  for (int i = 0; i < 4; ++i) {
    w1t[i] = (bf16_t*)alloc((size_t)FFD * DD * 2);
    w2t[i] = (bf16_t*)alloc((size_t)DD * FFD * 2);
  }
  float*  x     = (float*)alloc((size_t)MROWS * DD * 4);
  float*  y     = (float*)alloc((size_t)MROWS * DD * 4);
  float*  qb    = (float*)alloc((size_t)MROWS * DD * 4);
  float*  kb    = (float*)alloc((size_t)MROWS * DD * 4);
  float*  vb    = (float*)alloc((size_t)MROWS * DD * 4);
  float*  att   = (float*)alloc((size_t)MROWS * DD * 4);
  float*  f1    = (float*)alloc((size_t)MROWS * FFD * 4);
  bf16_t* hbf   = (bf16_t*)alloc((size_t)MROWS * DD * 2);
  bf16_t* attbf = (bf16_t*)alloc((size_t)MROWS * DD * 2);
  bf16_t* f1bf  = (bf16_t*)alloc((size_t)MROWS * FFD * 2);

  // ---- param indices (sorted-key pytree flatten) ----
  // attn blocks: 0=enc0.sa 1=enc1.sa 2=dec0.sa 3=dec0.ca 4=dec1.sa 5=dec1.ca
  const int A_pb[6] = {54, 67, 19,  2, 41, 24};
  const int A_pw[6] = {55, 68, 20,  3, 42, 25};
  const int A_wk[6] = {56, 69, 21,  4, 43, 26};
  const int A_wq[6] = {57, 70, 22,  5, 44, 27};
  const int A_wv[6] = {58, 71, 23,  6, 45, 28};
  // ff blocks: 0=enc0 1=enc1 2=dec0 3=dec1
  const int FB1[4] = {46, 59,  7, 29};
  const int FB2[4] = {47, 60,  8, 30};
  const int FW1[4] = {48, 61,  9, 31};
  const int FW2[4] = {49, 62, 10, 32};
  const int I_LMB = 72, I_LMW = 73, I_LNFB = 74, I_LNFG = 75;
  const int I_POS = 76, I_TOKIN = 77, I_TOKOUT = 78;

  // ---- weight prep ----
  for (int i = 0; i < 6; ++i) {
    k_conv_attn_w<<<4096 / 256, 256, 0, stream>>>(F(A_wq[i]), qwt[i]);
    k_conv_attn_w<<<4096 / 256, 256, 0, stream>>>(F(A_wk[i]), kwt[i]);
    k_conv_attn_w<<<4096 / 256, 256, 0, stream>>>(F(A_wv[i]), vwt[i]);
    k_conv_transpose<<<4096 / 256, 256, 0, stream>>>(F(A_pw[i]), pwt[i], DD, DD);
  }
  for (int i = 0; i < 4; ++i) {
    k_conv_transpose<<<(DD * FFD) / 256, 256, 0, stream>>>(F(FW1[i]), w1t[i], DD, FFD);
    k_conv_transpose<<<(DD * FFD) / 256, 256, 0, stream>>>(F(FW2[i]), w2t[i], FFD, DD);
  }
  k_conv_transpose<<<(DD * VOUTD) / 256, 256, 0, stream>>>(F(I_LMW), lmwt, DD, VOUTD);

  auto gemm = [&](const bf16_t* A, const bf16_t* Bt, const float* bias,
                  const float* res, float* C, bf16_t* Cbf, int N, int K, int relu) {
    dim3 g(MROWS / 16, N / 64);
    k_gemm<<<g, 32, 0, stream>>>(A, Bt, bias, res, C, Cbf, MROWS, N, K, relu);
  };
  auto ln = [&](const float* xin, const float* g_, const float* b_, bf16_t* o) {
    k_ln_bf16<<<MROWS / 256, 256, 0, stream>>>(xin, g_, b_, o);
  };
  auto attn = [&](int causal) {
    k_attn<<<(BB * HH * TT) / 256, 256, 0, stream>>>(qb, kb, vb, att, attbf, causal);
  };

  // ---- encoder ----
  k_embed_enc<<<(MROWS * DD) / 256, 256, 0, stream>>>(idx, F(I_TOKIN), F(I_POS), x);
  for (int l = 0; l < 2; ++l) {
    int blk = l;                                        // enc self-attn block
    ln(x, F(51 + 13 * l), F(50 + 13 * l), hbf);         // ln1
    gemm(hbf, qwt[blk], nullptr, nullptr, qb, nullptr, DD, DD, 0);
    gemm(hbf, kwt[blk], nullptr, nullptr, kb, nullptr, DD, DD, 0);
    gemm(hbf, vwt[blk], nullptr, nullptr, vb, nullptr, DD, DD, 0);
    attn(0);
    gemm(attbf, pwt[blk], F(A_pb[blk]), x, x, nullptr, DD, DD, 0);
    ln(x, F(53 + 13 * l), F(52 + 13 * l), hbf);         // ln2
    gemm(hbf, w1t[l], F(FB1[l]), nullptr, f1, f1bf, FFD, DD, 1);
    gemm(f1bf, w2t[l], F(FB2[l]), x, x, nullptr, DD, FFD, 0);
  }
  // x == enc_out

  // ---- decoder ----
  k_embed_dec<<<(MROWS * DD) / 256, 256, 0, stream>>>(tgt, F(I_TOKOUT), F(I_POS), y);
  for (int l = 0; l < 2; ++l) {
    const int bl = 2 + 22 * l;
    const int sa = 2 + 2 * l, ca = 3 + 2 * l;
    // causal self-attention
    ln(y, F(bl + 10), F(bl + 9), hbf);                  // ln1
    gemm(hbf, qwt[sa], nullptr, nullptr, qb, nullptr, DD, DD, 0);
    gemm(hbf, kwt[sa], nullptr, nullptr, kb, nullptr, DD, DD, 0);
    gemm(hbf, vwt[sa], nullptr, nullptr, vb, nullptr, DD, DD, 0);
    attn(1);
    gemm(attbf, pwt[sa], F(A_pb[sa]), y, y, nullptr, DD, DD, 0);
    // cross-attention: q from ln3(y), k/v from ln2(enc_out)
    ln(y, F(bl + 14), F(bl + 13), hbf);                 // ln3
    gemm(hbf, qwt[ca], nullptr, nullptr, qb, nullptr, DD, DD, 0);
    ln(x, F(bl + 12), F(bl + 11), hbf);                 // ln2 on enc_out
    gemm(hbf, kwt[ca], nullptr, nullptr, kb, nullptr, DD, DD, 0);
    gemm(hbf, vwt[ca], nullptr, nullptr, vb, nullptr, DD, DD, 0);
    attn(0);
    gemm(attbf, pwt[ca], F(A_pb[ca]), y, y, nullptr, DD, DD, 0);
    // feed-forward
    ln(y, F(bl + 16), F(bl + 15), hbf);                 // ln4
    const int fb = 2 + l;
    gemm(hbf, w1t[fb], F(FB1[fb]), nullptr, f1, f1bf, FFD, DD, 1);
    gemm(f1bf, w2t[fb], F(FB2[fb]), y, y, nullptr, DD, FFD, 0);
  }

  // ---- final LN + LM head (write-bandwidth bound: 327 MB f32 out) ----
  ln(y, F(I_LNFG), F(I_LNFB), hbf);
  gemm(hbf, lmwt, F(I_LMB), nullptr, (float*)d_out, nullptr, VOUTD, DD, 0);
}
